// StabilizedSNNAdapter_68899865362574
// MI455X (gfx1250) — matrix operand order repaired and checked by reading, not compiled, not run on hardware
//
#include <hip/hip_runtime.h>

// ---------------------------------------------------------------------------
// CDNA5 (gfx1250) implementation of StabilizedSNNAdapter
//   LN -> GEMM(bf16 WMMA, async global->LDS double buffering) ->
//   fused global-norm -> LIF scan (prefetched) -> LN -> GEMM -> residual
// ---------------------------------------------------------------------------

typedef __attribute__((ext_vector_type(16))) __bf16 v16bf;
typedef __attribute__((ext_vector_type(8)))  float  v8f;
typedef __attribute__((ext_vector_type(4)))  __bf16 v4bf;

#define LN_EPS    1e-5f
#define NORM_CLIP 50.0f
#define DECAY     0.25f

// CDNA5 async global->LDS copies (ASYNCcnt-tracked, no data VGPRs).
#if defined(__AMDGCN__) && __has_builtin(__builtin_amdgcn_global_load_async_to_lds_b128)
#define USE_ASYNC_LDS 1
#else
#define USE_ASYNC_LDS 0
#endif

#if USE_ASYNC_LDS
// Builtin signature (from hipcc diagnostic): first arg is
//   '__vector_size__(16) int __device__ *'  (global int4*),
// second is the LDS-side int4*, then Ii offset, Ii cpol.
typedef int v4i_vs __attribute__((__vector_size__(16)));

__device__ __forceinline__ void async_copy_b128(const void* gsrc, void* ldst) {
  typedef __attribute__((address_space(1))) v4i_vs* gp_t;
  typedef __attribute__((address_space(3))) v4i_vs* lp_t;
  __builtin_amdgcn_global_load_async_to_lds_b128((gp_t)gsrc, (lp_t)ldst, 0, 0);
}
__device__ __forceinline__ void wait_async0() {
#if __has_builtin(__builtin_amdgcn_s_wait_asynccnt)
  __builtin_amdgcn_s_wait_asynccnt(0);
#else
  asm volatile("s_wait_asynccnt 0x0" ::: "memory");
#endif
}
#endif

// ---------------------------------------------------------------------------
// block-wide sum for 256-thread blocks (wave32: 8 waves)
// ---------------------------------------------------------------------------
__device__ __forceinline__ float block_reduce_sum_256(float v, float* red) {
  #pragma unroll
  for (int off = 16; off > 0; off >>= 1) v += __shfl_down(v, off, 32);
  const int lane = threadIdx.x & 31;
  const int wave = threadIdx.x >> 5;
  if (lane == 0) red[wave] = v;
  __syncthreads();
  if (wave == 0) {
    float s = (lane < 8) ? red[lane] : 0.f;
    #pragma unroll
    for (int off = 4; off > 0; off >>= 1) s += __shfl_down(s, off, 32);
    if (lane == 0) red[0] = s;
  }
  __syncthreads();
  return red[0];
}

// ---------------------------------------------------------------------------
// fp32 -> bf16 weight conversion (n4 = n/4 float4 units)
// ---------------------------------------------------------------------------
__global__ void f32_to_bf16_kernel(const float* __restrict__ in,
                                   __bf16* __restrict__ out, int n4) {
  int i = blockIdx.x * blockDim.x + threadIdx.x;
  if (i < n4) {
    float4 v = ((const float4*)in)[i];
    v4bf o;
    o[0] = (__bf16)v.x; o[1] = (__bf16)v.y; o[2] = (__bf16)v.z; o[3] = (__bf16)v.w;
    ((v4bf*)out)[i] = o;
  }
}

// ---------------------------------------------------------------------------
// LayerNorm over last dim (H = 1024, one 256-thread block per row), bf16 out
// ---------------------------------------------------------------------------
template <typename T>
__global__ __launch_bounds__(256) void ln_to_bf16_kernel(
    const T* __restrict__ x, const float* __restrict__ gamma,
    const float* __restrict__ beta, __bf16* __restrict__ out, int H) {
  __shared__ float red0[8];
  __shared__ float red1[8];
  const int row  = blockIdx.x;
  const int tid  = threadIdx.x;
  const size_t rb = (size_t)row * H;
  const int base  = tid * 4;

  float v0 = (float)x[rb + base + 0];
  float v1 = (float)x[rb + base + 1];
  float v2 = (float)x[rb + base + 2];
  float v3 = (float)x[rb + base + 3];

  float sum  = block_reduce_sum_256(v0 + v1 + v2 + v3, red0);
  float sum2 = block_reduce_sum_256(v0*v0 + v1*v1 + v2*v2 + v3*v3, red1);

  float mu  = sum / (float)H;
  float var = sum2 / (float)H - mu * mu;
  float inv = rsqrtf(var + LN_EPS);

  v4bf o;
  o[0] = (__bf16)((v0 - mu) * inv * gamma[base + 0] + beta[base + 0]);
  o[1] = (__bf16)((v1 - mu) * inv * gamma[base + 1] + beta[base + 1]);
  o[2] = (__bf16)((v2 - mu) * inv * gamma[base + 2] + beta[base + 2]);
  o[3] = (__bf16)((v3 - mu) * inv * gamma[base + 3] + beta[base + 3]);
  *(v4bf*)(out + rb + base) = o;
}

// ---------------------------------------------------------------------------
// bf16 WMMA GEMM:  out[m][n] = sum_k A[m][k] * W[n][k]  (+ epilogue)
//   block tile 256(M) x 64(N), K-chunk 32; 8 waves arranged 4(M) x 2(N),
//   each wave computes 64x32 via 4x2 v_wmma_f32_16x16x32_bf16 accumulators.
//   Double-buffered LDS; next chunk streamed with GLOBAL_LOAD_ASYNC_TO_LDS
//   (no data VGPRs, ASYNCcnt) while wmma runs on the current buffer.
// MODE 1: out = acc + bias (fp32), fused block sum-of-squares -> part[]
// MODE 2: out = resid + clip(a)*clip(g)*(acc + bias)
// ---------------------------------------------------------------------------
template <int MODE>
__global__ __launch_bounds__(256) void gemm_bf16_kernel(
    const __bf16* __restrict__ A,      // [M][K]
    const __bf16* __restrict__ W,      // [N][K]
    const float*  __restrict__ bias,   // [N]
    const float*  __restrict__ resid,  // [M][N] (MODE 2)
    const float*  __restrict__ alpha_p,
    const float*  __restrict__ gs_p,
    float* __restrict__ out,
    float* __restrict__ part,          // [gridDim.x*gridDim.y] (MODE 1)
    int M, int N, int K) {
  // padded stride 40 elems = 80B = 20 banks -> conflict-free b128 frag reads
  __shared__ __align__(16) __bf16 As[2][256 * 40];
  __shared__ __align__(16) __bf16 Bs[2][64 * 40];
  __shared__ float red[8];

  const int tid  = threadIdx.x;
  const int lane = tid & 31;
  const int wave = tid >> 5;
  const int wm   = wave & 3;   // 64-row slice within block tile
  const int wn   = wave >> 2;  // 32-col slice within block tile
  const int mBase = blockIdx.y * 256;
  const int nBase = blockIdx.x * 64;

  const int l16  = lane & 15;
  const int half = lane >> 4;  // 0: lanes 0-15, 1: lanes 16-31
  const int aRow = tid >> 2, aCu = tid & 3;

  v8f acc[4][2];
  #pragma unroll
  for (int am = 0; am < 4; ++am)
    #pragma unroll
    for (int bn = 0; bn < 2; ++bn)
      #pragma unroll
      for (int r = 0; r < 8; ++r) acc[am][bn][r] = 0.f;

  // ---- compute on current LDS buffer: 6 frag loads, 8 wmma ----
  auto compute = [&](int b) {
    v16bf afr[4], bfr[2];
    #pragma unroll
    for (int am = 0; am < 4; ++am) {
      int r = wm * 64 + am * 16 + l16;
      int c = half * 8;  // lanes 0-15: K 0-7/16-23; lanes 16-31: K 8-15/24-31
      union { uint4 u[2]; v16bf v; } cv;
      cv.u[0] = *(const uint4*)(&As[b][r * 40 + c]);
      cv.u[1] = *(const uint4*)(&As[b][r * 40 + c + 16]);
      afr[am] = cv.v;
    }
    #pragma unroll
    for (int bn = 0; bn < 2; ++bn) {
      int n = wn * 32 + bn * 16 + l16;
      int c = half * 8;  // lanes 0-15: K 0-7/16-23; lanes 16-31: K 8-15/24-31
      union { uint4 u[2]; v16bf v; } cv;
      cv.u[0] = *(const uint4*)(&Bs[b][n * 40 + c]);
      cv.u[1] = *(const uint4*)(&Bs[b][n * 40 + c + 16]);
      bfr[bn] = cv.v;
    }
    #pragma unroll
    for (int am = 0; am < 4; ++am)
      #pragma unroll
      for (int bn = 0; bn < 2; ++bn)
        acc[am][bn] = __builtin_amdgcn_wmma_f32_16x16x32_bf16(
            false, afr[am], false, bfr[bn], (short)0, acc[am][bn], false, false);
  };

  const int nch = K / 32;
  int buf = 0;

#if USE_ASYNC_LDS
  // ---- stage K-chunk kk into LDS buffer b: 5 async b128 per thread ----
  auto stage = [&](int kk, int b) {
    #pragma unroll
    for (int c = 0; c < 4; ++c) {
      int u = tid + 256 * c, row = u >> 2, cu = u & 3;
      async_copy_b128(A + (size_t)(mBase + row) * K + kk + cu * 8,
                      &As[b][row * 40 + cu * 8]);
    }
    async_copy_b128(W + (size_t)(nBase + aRow) * K + kk + aCu * 8,
                    &Bs[b][aRow * 40 + aCu * 8]);
  };

  stage(0, 0);
  wait_async0();
  __syncthreads();
  for (int i = 0; i < nch; ++i) {
    // LDS[buf^1] is free here: all reads of it finished before the last
    // barrier (wmma forced dscnt==0 before each wave signaled it).
    if (i + 1 < nch) stage((i + 1) * 32, buf ^ 1);
    compute(buf);
    if (i + 1 < nch) {
      wait_async0();
      __syncthreads();
      buf ^= 1;
    }
  }
#else
  // ---- fallback: stage through registers (sync loads + ds stores) ----
  uint4 pa[4], pb;
  auto load_global = [&](int kk) {
    #pragma unroll
    for (int c = 0; c < 4; ++c) {
      int u = tid + 256 * c, row = u >> 2, cu = u & 3;
      pa[c] = *(const uint4*)(A + (size_t)(mBase + row) * K + kk + cu * 8);
    }
    pb = *(const uint4*)(W + (size_t)(nBase + aRow) * K + kk + aCu * 8);
  };
  auto store_lds = [&](int b) {
    #pragma unroll
    for (int c = 0; c < 4; ++c) {
      int u = tid + 256 * c, row = u >> 2, cu = u & 3;
      *(uint4*)(&As[b][row * 40 + cu * 8]) = pa[c];
    }
    *(uint4*)(&Bs[b][aRow * 40 + aCu * 8]) = pb;
  };

  load_global(0);
  store_lds(0);
  __syncthreads();
  for (int i = 0; i < nch; ++i) {
    if (i + 1 < nch) load_global((i + 1) * 32);
    compute(buf);
    if (i + 1 < nch) {
      store_lds(buf ^ 1);
      __syncthreads();
      buf ^= 1;
    }
  }
#endif

  float ascale = 1.f;
  if (MODE == 2) {
    float a = fminf(fmaxf(alpha_p[0], 0.0f), 0.5f);
    float g = fminf(fmaxf(gs_p[0], 0.1f), 1.0f);
    ascale = a * g;
  }

  // C/D layout: VGPR r -> M = r + 8*half, N = lane%16
  float ss = 0.f;
  #pragma unroll
  for (int am = 0; am < 4; ++am) {
    #pragma unroll
    for (int bn = 0; bn < 2; ++bn) {
      int col = nBase + wn * 32 + bn * 16 + l16;
      float b = bias[col];
      #pragma unroll
      for (int r = 0; r < 8; ++r) {
        int row  = mBase + wm * 64 + am * 16 + half * 8 + r;
        size_t o = (size_t)row * N + col;
        float v  = acc[am][bn][r] + b;
        if (MODE == 1) { out[o] = v; ss += v * v; }
        else           { out[o] = resid[o] + ascale * v; }
      }
    }
  }
  if (MODE == 1) {
    __syncthreads();
    float t = block_reduce_sum_256(ss, red);
    if (tid == 0) part[blockIdx.y * gridDim.x + blockIdx.x] = t;
  }
}

// 1024 block partials -> scale = (||x|| > 50) ? 50/||x|| : 1   (deterministic)
__global__ __launch_bounds__(256) void norm_scale_kernel(
    const float* __restrict__ part, float* __restrict__ scale_out) {
  __shared__ float red[8];
  int tid = threadIdx.x;
  float s = part[tid] + part[tid + 256] + part[tid + 512] + part[tid + 768];
  float t = block_reduce_sum_256(s, red);
  if (tid == 0) {
    float n = sqrtf(t);
    scale_out[0] = (n > NORM_CLIP) ? (NORM_CLIP / n) : 1.0f;
  }
}

// ---------------------------------------------------------------------------
// LIF scan: one thread per (b,h) channel, sequential over t.
// Separate in/out buffers (__restrict__) so unrolled loads pipeline ahead of
// the dependent mem-update chain; explicit prefetch (global_prefetch_b8)
// keeps the stream ahead in cache. Spikes (0..4) stored exactly in bf16.
// ---------------------------------------------------------------------------
__global__ __launch_bounds__(256) void snn_scan_kernel(
    const float* __restrict__ xp,      // [B*S][H] fp32 (x_proj)
    __bf16* __restrict__ sp,           // [B*S][H] bf16 spikes
    const float* __restrict__ scale_p, const float* __restrict__ thre_p,
    int S, int H) {
  int idx = blockIdx.x * blockDim.x + threadIdx.x;  // 0 .. B*H-1
  int b = idx / H, h = idx % H;
  float scale = scale_p[0];
  float th = thre_p[0];
  float t2 = 2.f * th, t3 = 3.f * th, t4 = 4.f * th;
  float mem = 0.f;
  size_t base = (size_t)b * S * H + h;
  #pragma unroll 8
  for (int t = 0; t < S; ++t) {
    size_t a = base + (size_t)t * H;
    int tp = (t + 16 < S) ? (t + 16) : (S - 1);   // clamped prefetch distance
    __builtin_prefetch(xp + base + (size_t)tp * H, 0, 0);
    float u = xp[a] * scale;
    mem = mem * DECAY + u;
    float spike = (mem > th ? 1.f : 0.f) + (mem > t2 ? 1.f : 0.f) +
                  (mem > t3 ? 1.f : 0.f) + (mem > t4 ? 1.f : 0.f);
    mem -= spike * th;
    sp[a] = (__bf16)spike;
  }
}

// ---------------------------------------------------------------------------
// host launcher
// ---------------------------------------------------------------------------
extern "C" void kernel_launch(void* const* d_in, const int* in_sizes, int n_in,
                              void* d_out, int out_size, void* d_ws, size_t ws_size,
                              hipStream_t stream) {
  const float* x        = (const float*)d_in[0];
  const float* in_gamma = (const float*)d_in[1];
  const float* in_beta  = (const float*)d_in[2];
  const float* preW     = (const float*)d_in[3];
  const float* preb     = (const float*)d_in[4];
  const float* pn_gamma = (const float*)d_in[5];
  const float* pn_beta  = (const float*)d_in[6];
  const float* postW    = (const float*)d_in[7];
  const float* postb    = (const float*)d_in[8];
  const float* alpha_p  = (const float*)d_in[9];
  const float* gs_p     = (const float*)d_in[10];
  const float* thre_p   = (const float*)d_in[11];

  const int H = in_sizes[1];          // 1024
  const int M = in_sizes[0] / H;      // B*S = 16384
  const int B = 4;                    // per reference setup
  const int S = M / B;                // 4096
  const int K = H, N = H;

  // workspace layout
  char* ws = (char*)d_ws;
  size_t off = 0;
  __bf16* Wpre_bf  = (__bf16*)(ws + off); off += (size_t)N * K * sizeof(__bf16);
  __bf16* Wpost_bf = (__bf16*)(ws + off); off += (size_t)N * K * sizeof(__bf16);
  __bf16* Abf      = (__bf16*)(ws + off); off += (size_t)M * K * sizeof(__bf16);  // LN outs
  float*  xproj    = (float*)(ws + off);  off += (size_t)M * N * sizeof(float);
  __bf16* spikes   = (__bf16*)(ws + off); off += (size_t)M * N * sizeof(__bf16);
  float*  partials = (float*)(ws + off);  off += 1024 * sizeof(float);
  float*  scale_ws = (float*)(ws + off);  off += sizeof(float);
  (void)ws_size; (void)n_in;

  float* out = (float*)d_out;
  const dim3 blk(256);
  const dim3 gemm_grid(N / 64, M / 256);   // 16 x 64 = 1024 blocks

  // 1) weights -> bf16
  f32_to_bf16_kernel<<<(N * K / 4 + 255) / 256, blk, 0, stream>>>(preW, Wpre_bf, N * K / 4);
  f32_to_bf16_kernel<<<(N * K / 4 + 255) / 256, blk, 0, stream>>>(postW, Wpost_bf, N * K / 4);

  // 2) LN(x) -> bf16
  ln_to_bf16_kernel<float><<<M, blk, 0, stream>>>(x, in_gamma, in_beta, Abf, H);

  // 3) GEMM1: x_proj = LN(x) @ preW^T + preb, fused sum-of-squares partials
  gemm_bf16_kernel<1><<<gemm_grid, blk, 0, stream>>>(
      Abf, Wpre_bf, preb, nullptr, nullptr, nullptr, xproj, partials, M, N, K);

  // 4) deterministic global-norm clip scale (1024 partials -> 1)
  norm_scale_kernel<<<1, blk, 0, stream>>>(partials, scale_ws);

  // 5) LIF multi-threshold scan (soft reset), spikes -> bf16
  snn_scan_kernel<<<(B * H) / 256, blk, 0, stream>>>(xproj, spikes, scale_ws, thre_p, S, H);

  // 6) LN(spikes) -> bf16 (reuse Abf)
  ln_to_bf16_kernel<__bf16><<<M, blk, 0, stream>>>(spikes, pn_gamma, pn_beta, Abf, H);

  // 7) GEMM2 + residual: out = x + clip(a)*clip(g)*(LN(spikes) @ postW^T + postb)
  gemm_bf16_kernel<2><<<gemm_grid, blk, 0, stream>>>(
      Abf, Wpost_bf, postb, x, alpha_p, gs_p, out, nullptr, M, N, K);

  (void)out_size;
}
